// MGM3_unsup_37263136260678
// MI455X (gfx1250) — matrix operand order repaired and checked by reading, not compiled, not run on hardware
//
#include <hip/hip_runtime.h>
#include <math.h>

typedef unsigned short ushort_t;
typedef __attribute__((ext_vector_type(16))) __bf16 v16bf;
typedef __attribute__((ext_vector_type(8)))  float  v8f;

union V16 { uint4 q[2]; v16bf v; };

__device__ __forceinline__ ushort_t f2bf(float x) {
    unsigned u = __builtin_bit_cast(unsigned, x);
    unsigned r = (u + 0x7FFFu + ((u >> 16) & 1u)) >> 16;   // round-to-nearest-even
    return (ushort_t)r;
}

// byte offset of a generic pointer inside LDS (addrspacecast -> ptrtoint)
__device__ __forceinline__ unsigned lds_off(void* p) {
    return (unsigned)(unsigned long long)(__attribute__((address_space(3))) void*)p;
}

// ---------------------------------------------------------------------------
// GEMM:  C = Aop(MxK) @ B(KxN);  A is bf16 row-major, B given as Bt = B^T
// (NxK, bf16, row-major).  Output written TRANSPOSED: Ct (NxM), f32 or bf16.
// Block tile 128(M) x 64(N), 8 waves as 4x2, wave tile 32x32, K-step 32.
// Staging: GLOBAL_LOAD_ASYNC_TO_LDS_B128, TRIPLE-buffered with lookahead-2,
// so the bottom-of-step wait (asynccnt<=3) only needs the tile issued a full
// step earlier; the newest tile stays in flight across the barrier.
// ---------------------------------------------------------------------------
template<bool OUT_BF16>
__global__ __launch_bounds__(256)
void gemm_wmma(const ushort_t* __restrict__ Aop, const ushort_t* __restrict__ Bt,
               void* __restrict__ Ct, int ldA, int ldBt, int ldCt,
               int Kc, unsigned long long ctChunkStride)
{
    __shared__ __align__(16) ushort_t As[3][128 * 40];   // rows padded 32->40
    __shared__ __align__(16) ushort_t Bs[3][64 * 40];

    const int t     = threadIdx.x;
    const int m0    = blockIdx.x * 128;
    const int n0    = blockIdx.y * 64;
    const int kz    = blockIdx.z;
    const int kbase = kz * Kc;

    const int lane = t & 31, wave = t >> 5;
    const int wm = wave & 3, wn = wave >> 2;
    const int lr = lane & 15, h = lane >> 4;

    // staging assignments: A tile 128x32 (16 elems/thread), B tile 64x32 (8/thread)
    const int ar = t >> 1, ac = (t & 1) * 16;
    const int br = t >> 2, bc = (t & 3) * 8;

    const unsigned aLb = lds_off(&As[0][ar * 40 + ac]);
    const unsigned bLb = lds_off(&Bs[0][br * 40 + bc]);
    const ushort_t* aG = Aop + (size_t)(m0 + ar) * ldA + ac;
    const ushort_t* bG = Bt  + (size_t)(n0 + br) * ldBt + bc;

    v8f acc[2][2];
    const v8f zv = {0.f,0.f,0.f,0.f,0.f,0.f,0.f,0.f};
#pragma unroll
    for (int i = 0; i < 2; ++i)
#pragma unroll
        for (int j = 0; j < 2; ++j) acc[i][j] = zv;

    const int steps = Kc >> 5;

    auto issue_tile = [&](int buf, int kk) {
        unsigned aL = aLb + (unsigned)buf * (128u * 40u * 2u);
        unsigned bL = bLb + (unsigned)buf * (64u * 40u * 2u);
        unsigned long long ag = (unsigned long long)(const void*)(aG + kk);
        unsigned long long bg = (unsigned long long)(const void*)(bG + kk);
        asm volatile("global_load_async_to_lds_b128 %0, %1, off"
                     :: "v"(aL), "v"(ag) : "memory");
        asm volatile("global_load_async_to_lds_b128 %0, %1, off offset:16"
                     :: "v"(aL), "v"(ag) : "memory");   // offset applies to LDS+mem
        asm volatile("global_load_async_to_lds_b128 %0, %1, off"
                     :: "v"(bL), "v"(bg) : "memory");
    };

    // prologue: two tiles in flight, require only tile 0 to have landed
    issue_tile(0, kbase);
    if (steps > 1) {
        issue_tile(1, kbase + 32);
        asm volatile("s_wait_asynccnt 0x3" ::: "memory");
    } else {
        asm volatile("s_wait_asynccnt 0x0" ::: "memory");
    }
    __syncthreads();

    int cur = 0, pre = 2;
    for (int kt = 0; kt < steps; ++kt) {
        const bool deep = (kt + 2) < steps;
        if (deep) issue_tile(pre, kbase + (kt + 2) * 32);
        if (kt + 3 < steps)   // emits global_prefetch_b8
            __builtin_prefetch(aG + kbase + (kt + 3) * 32, 0, 3);

        const ushort_t* asb = As[cur];
        const ushort_t* bsb = Bs[cur];

        V16 af[2], bq[2];
#pragma unroll
        for (int mt = 0; mt < 2; ++mt) {
            const ushort_t* p = asb + (wm * 32 + mt * 16 + lr) * 40 + h * 8;
            af[mt].q[0] = *(const uint4*)p;          // K = h*8 .. +8
            af[mt].q[1] = *(const uint4*)(p + 16);   // K = 16+h*8 .. +8
        }
#pragma unroll
        for (int nt = 0; nt < 2; ++nt) {
            const ushort_t* p = bsb + (wn * 32 + nt * 16 + lr) * 40 + h * 8;
            bq[nt].q[0] = *(const uint4*)p;          // K = h*8 .. +8
            bq[nt].q[1] = *(const uint4*)(p + 16);   // K = 16+h*8 .. +8
        }
#pragma unroll
        for (int mt = 0; mt < 2; ++mt)
#pragma unroll
            for (int nt = 0; nt < 2; ++nt)
                acc[mt][nt] = __builtin_amdgcn_wmma_f32_16x16x32_bf16(
                    false, af[mt].v, false, bq[nt].v, (short)0, acc[mt][nt], false, false);

        // need tile kt+1 (issued a full step ago); tile kt+2 may stay in flight
        if (deep) asm volatile("s_wait_asynccnt 0x3" ::: "memory");
        else      asm volatile("s_wait_asynccnt 0x0" ::: "memory");
        __syncthreads();

        cur = (cur == 2) ? 0 : cur + 1;
        pre = (pre == 2) ? 0 : pre + 1;
    }

    // transposed epilogue: per lane 8 contiguous M values -> b128 stores
    char* cbase = (char*)Ct + (size_t)kz * ctChunkStride;
#pragma unroll
    for (int nt = 0; nt < 2; ++nt) {
        const int n = n0 + wn * 32 + nt * 16 + lr;
#pragma unroll
        for (int mt = 0; mt < 2; ++mt) {
            const int mb = m0 + wm * 32 + mt * 16 + h * 8;
            if constexpr (OUT_BF16) {
                ushort_t tmp[8];
#pragma unroll
                for (int r = 0; r < 8; ++r) tmp[r] = f2bf(acc[mt][nt][r]);
                *(uint4*)((ushort_t*)cbase + (size_t)n * ldCt + mb) = *(const uint4*)tmp;
            } else {
                float* dst = (float*)cbase + (size_t)n * ldCt + mb;
                float4 lo, hi;
                lo.x = acc[mt][nt][0]; lo.y = acc[mt][nt][1]; lo.z = acc[mt][nt][2]; lo.w = acc[mt][nt][3];
                hi.x = acc[mt][nt][4]; hi.y = acc[mt][nt][5]; hi.z = acc[mt][nt][6]; hi.w = acc[mt][nt][7];
                ((float4*)dst)[0] = lo;
                ((float4*)dst)[1] = hi;
            }
        }
    }
}

// ---------------------------------------------------------------------------
// one-shot elementwise f32 -> bf16 (for constant A, W), 4 elems/thread
// ---------------------------------------------------------------------------
__global__ __launch_bounds__(256)
void cvt_f32_bf16(const float* __restrict__ in, ushort_t* __restrict__ out)
{
    const size_t i4 = (size_t)blockIdx.x * 256 + threadIdx.x;
    float4 v = ((const float4*)in)[i4];
    ushort_t tmp[4] = { f2bf(v.x), f2bf(v.y), f2bf(v.z), f2bf(v.w) };
    ((uint2*)out)[i4] = *(const uint2*)tmp;
}

// U (4096x256 f32) -> Ut (256x4096 bf16)
__global__ __launch_bounds__(256)
void transpose_cvt_f32bf(const float* __restrict__ in, ushort_t* __restrict__ out)
{
    __shared__ float tile[32][33];
    const int m0 = blockIdx.x * 32, n0 = blockIdx.y * 32;
    const int tx = threadIdx.x, ty = threadIdx.y;
#pragma unroll
    for (int k = 0; k < 4; ++k)
        tile[ty + 8 * k][tx] = in[(size_t)(m0 + ty + 8 * k) * 256 + n0 + tx];
    __syncthreads();
#pragma unroll
    for (int k = 0; k < 4; ++k)
        out[(size_t)(n0 + ty + 8 * k) * 4096 + m0 + tx] = f2bf(tile[tx][ty + 8 * k]);
}

// AUt (256x4096 bf16) -> AU (4096x256 bf16)
__global__ __launch_bounds__(256)
void transpose_bf(const ushort_t* __restrict__ in, ushort_t* __restrict__ out)
{
    __shared__ ushort_t tile[32][33];
    const int m0 = blockIdx.x * 32, n0 = blockIdx.y * 32;
    const int tx = threadIdx.x, ty = threadIdx.y;
#pragma unroll
    for (int k = 0; k < 4; ++k)
        tile[ty + 8 * k][tx] = in[(size_t)(n0 + ty + 8 * k) * 4096 + m0 + tx];
    __syncthreads();
#pragma unroll
    for (int k = 0; k < 4; ++k)
        out[(size_t)(m0 + ty + 8 * k) * 256 + n0 + tx] = tile[tx][ty + 8 * k];
}

// sum 8 split-K partials (each 256x256 f32) -> S^T bf16
__global__ __launch_bounds__(256)
void reduceS(const float* __restrict__ Sp, ushort_t* __restrict__ Sb)
{
    const int idx = blockIdx.x * 256 + threadIdx.x;
    float s = 0.f;
#pragma unroll
    for (int z = 0; z < 8; ++z) s += Sp[z * 65536 + idx];
    Sb[idx] = f2bf(s);
}

// ---------------------------------------------------------------------------
// Fused V-formation + 20-step log-Sinkhorn (dual form) + exp, per group.
// V/tau (256x256 f32) lives read-only in LDS (rows padded to 260 floats);
// only the duals f[256], g[256] are updated.  dynamic LDS = 268288 B.
// ---------------------------------------------------------------------------
__global__ __launch_bounds__(256)
void sinkhorn_kernel(const float* __restrict__ quadT, const float* __restrict__ wuT,
                     float* __restrict__ Uout, float tau)
{
    extern __shared__ float smem[];
    float* Vt = smem;                  // [256][260]
    float* fr = smem + 256 * 260;      // 256
    float* gc = fr + 256;              // 256

    const int t  = threadIdx.x;
    const int gr = blockIdx.x;
    const float sc = (1.0f / 16.0f) / tau;

    for (int idx = t; idx < 65536; idx += 256) {
        const int n = idx >> 8, m = idx & 255;
        const size_t src = (size_t)n * 4096 + (size_t)gr * 256 + m;
        Vt[m * 260 + n] = (quadT[src] + wuT[src]) * sc;
    }
    fr[t] = 0.0f;
    gc[t] = 0.0f;
    __syncthreads();

    for (int it = 0; it < 20; ++it) {
        if ((it & 1) == 0) {
            // f[t] = LSE_n( Vt[t][n] - g[n] )
            const float4* row = (const float4*)(Vt + t * 260);
            const float4* g4p = (const float4*)gc;
            float mx = -3.0e38f;
            for (int j = 0; j < 64; ++j) {
                float4 v = row[j], g4 = g4p[j];
                mx = fmaxf(mx, fmaxf(fmaxf(v.x - g4.x, v.y - g4.y),
                                     fmaxf(v.z - g4.z, v.w - g4.w)));
            }
            float s = 0.f;
            for (int j = 0; j < 64; ++j) {
                float4 v = row[j], g4 = g4p[j];
                s += __expf(v.x - g4.x - mx) + __expf(v.y - g4.y - mx)
                   + __expf(v.z - g4.z - mx) + __expf(v.w - g4.w - mx);
            }
            fr[t] = mx + __logf(s);
        } else {
            // g[t] = LSE_m( Vt[m][t] - f[m] )
            float mx = -3.0e38f;
            for (int i = 0; i < 256; ++i) mx = fmaxf(mx, Vt[i * 260 + t] - fr[i]);
            float s = 0.f;
            for (int i = 0; i < 256; ++i) s += __expf(Vt[i * 260 + t] - fr[i] - mx);
            gc[t] = mx + __logf(s);
        }
        __syncthreads();
    }

    for (int idx = t; idx < 65536; idx += 256) {
        const int m = idx >> 8, n = idx & 255;
        Uout[((size_t)gr * 256 + m) * 256 + n] = __expf(Vt[m * 260 + n] - fr[m] - gc[n]);
    }
}

// ---------------------------------------------------------------------------
extern "C" void kernel_launch(void* const* d_in, const int* in_sizes, int n_in,
                              void* d_out, int out_size, void* d_ws, size_t ws_size,
                              hipStream_t stream)
{
    const float* A  = (const float*)d_in[0];   // 4096x4096
    const float* W  = (const float*)d_in[1];   // 4096x4096
    const float* U0 = (const float*)d_in[2];   // 4096x256
    float* Uf = (float*)d_out;                 // 4096x256, also iteration state

    char* ws = (char*)d_ws;
    ushort_t* Utbf  = (ushort_t*)(ws);                          // 2 MB  (256x4096 bf16)
    ushort_t* AUtbf = (ushort_t*)(ws + (2ull << 20));           // 2 MB
    ushort_t* AUbf  = (ushort_t*)(ws + (4ull << 20));           // 2 MB  (4096x256 bf16)
    float*    Spart = (float*)   (ws + (6ull << 20));           // 2 MB  (8 x 256x256 f32)
    ushort_t* Sbf   = (ushort_t*)(ws + (8ull << 20));           // 128 KB (S^T bf16)
    float*    quadT = (float*)   (ws + (9ull << 20));           // 4 MB  (256x4096 f32)
    float*    wuT   = (float*)   (ws + (13ull << 20));          // 4 MB
    ushort_t* Abf   = (ushort_t*)(ws + (17ull << 20));          // 32 MB (4096x4096 bf16)
    ushort_t* Wbf   = (ushort_t*)(ws + (49ull << 20));          // 32 MB

    // one-shot: constant operands to bf16 (A, W reused by all 8 iterations)
    cvt_f32_bf16<<<16384, 256, 0, stream>>>(A, Abf);
    cvt_f32_bf16<<<16384, 256, 0, stream>>>(W, Wbf);

    for (int i = 0; i < 8; ++i) {
        const float* Ucur = (i == 0) ? U0 : Uf;

        // 1. Utbf = bf16(U^T)
        transpose_cvt_f32bf<<<dim3(128, 8), dim3(32, 8), 0, stream>>>(Ucur, Utbf);

        // 2. AUt = (A @ U)^T  (bf16 out)
        gemm_wmma<true><<<dim3(32, 4, 1), 256, 0, stream>>>(
            Abf, Utbf, AUtbf, 4096, 4096, 4096, 4096, 0ull);

        // 3. AU = AUt^T
        transpose_bf<<<dim3(128, 8), dim3(32, 8), 0, stream>>>(AUtbf, AUbf);

        // 4. split-K partials of (U^T @ AU)^T
        gemm_wmma<false><<<dim3(2, 4, 8), 256, 0, stream>>>(
            Utbf, AUtbf, Spart, 4096, 4096, 256, 512, 262144ull);

        // 5. S^T = bf16(sum of partials)
        reduceS<<<256, 256, 0, stream>>>(Spart, Sbf);

        // 6. quadT = (AU @ S)^T  (f32 out)
        gemm_wmma<false><<<dim3(32, 4, 1), 256, 0, stream>>>(
            AUbf, Sbf, quadT, 256, 256, 4096, 256, 0ull);

        // 7. wuT = (W @ U)^T  (f32 out)
        gemm_wmma<false><<<dim3(32, 4, 1), 256, 0, stream>>>(
            Wbf, Utbf, wuT, 4096, 4096, 4096, 4096, 0ull);

        // 8. fused V + sinkhorn(dual) + exp -> U
        float taui = fmaxf(0.1f * powf(0.5f, (float)(i / 2)), 0.01f);
        sinkhorn_kernel<<<16, 256, 268288, stream>>>(quadT, wuT, Uf, taui);
    }
}
